// MMN_32031866094240
// MI455X (gfx1250) — matrix operand order repaired and checked by reading, not compiled, not run on hardware
//
#include <hip/hip_runtime.h>

typedef _Float16 f16;
typedef __attribute__((ext_vector_type(16))) _Float16 v16h;
typedef __attribute__((ext_vector_type(8)))  _Float16 v8h;
typedef __attribute__((ext_vector_type(8)))  float    v8f;

#define HW   3600
#define HWP  3712           // HW padded up to a multiple of 128 (29 tiles)
#define CK   3072           // concatenated K: 2048 (level4) + 1024 (level3)
#define BK   64             // K-chunk per stage
#define BT   128            // M/N tile per workgroup
#define LDT  72             // LDS row stride in halfs (64 + 8 pad, 16B aligned)

__device__ __forceinline__ v16h cat16(v8h lo, v8h hi) {
    return __builtin_shufflevector(lo, hi, 0,1,2,3,4,5,6,7,8,9,10,11,12,13,14,15);
}

// A-matrix 16x32 f16 fragment (ISA 7.12.2): m = lane&15, hl = lane>>4.
// hl 0 holds K {0..7, 16..23}; hl 1 holds K {8..15, 24..31}:
// two contiguous 16B chunks of a k-contiguous LDS row.
__device__ __forceinline__ v16h load_a_frag(const f16* As, int row, int hl) {
    v8h lo = *(const v8h*)(As + row * LDT + hl * 8);
    v8h hi = *(const v8h*)(As + row * LDT + 16 + hl * 8);
    return cat16(lo, hi);
}

// B-matrix 32x16 f16 fragment: n = lane&15; hl 0 holds K {0..7, 16..23}, hl 1 K {8..15, 24..31}.
__device__ __forceinline__ v16h load_b_frag(const f16* BsT, int n, int hl) {
    v8h lo = *(const v8h*)(BsT + n * LDT + hl * 8);
    v8h hi = *(const v8h*)(BsT + n * LDT + 16 + hl * 8);
    return cat16(lo, hi);
}

// ---------------------------------------------------------------------------
// L2-normalize over channels, transpose to [pixel][c] (k-contiguous rows) at
// column offset co of a CK-wide row, optional w_chan scale folded in.
// Zero-fills its column range for padded pixel rows [HW, HWP).
__global__ __launch_bounds__(256)
void normalize_t_kernel(const float* __restrict__ in, f16* __restrict__ out,
                        int C, int co, int nb,
                        const float* __restrict__ w_chan, int widx) {
    int p = blockIdx.x * 256 + threadIdx.x;
    if (p >= nb * HWP) return;
    int b = p / HWP, pix = p - b * HWP;
    f16* orow = out + (size_t)p * CK + co;
    if (pix >= HW) {
        for (int c = 0; c < C; ++c) orow[c] = (f16)0.f;
        return;
    }
    const float* base = in + (size_t)b * C * HW + pix;
    float sum = 0.f;
    for (int c = 0; c < C; ++c) { float v = base[(size_t)c * HW]; sum += v * v; }
    float inv = 1.0f / fmaxf(sqrtf(sum), 1e-12f);
    if (widx >= 0) inv *= w_chan[widx];
    for (int c = 0; c < C; ++c) orow[c] = (f16)(base[(size_t)c * HW] * inv);
}

// f32 (rows, HW) -> f16 (rows, HWP) with zero s-padding (padded K must be 0).
__global__ __launch_bounds__(256)
void convert_pad_kernel(const float* __restrict__ in, f16* __restrict__ out,
                        int rows) {
    int i = blockIdx.x * 256 + threadIdx.x;
    if (i >= rows * HWP) return;
    int r = i / HWP, s = i - r * HWP;
    out[i] = (s < HW) ? (f16)in[(size_t)r * HW + s] : (f16)0.f;
}

// ---------------------------------------------------------------------------
// corr[b,q,s] = TEMP * <qnT_q, snT_s>  over concatenated K=3072
// (w_chan already folded into qnT). 128x128 WG tile, 8 waves (2x4),
// each wave 64x32 (4x2 subtiles), K-chunks of 64.
__global__ __launch_bounds__(256)
void corr_kernel(const f16* __restrict__ qnT, const f16* __restrict__ snT,
                 float* __restrict__ corr) {
    __shared__ f16 As[BT * LDT];   // [q_local][k]
    __shared__ f16 Bs[BT * LDT];   // [s_local][k]

    const int q0 = blockIdx.x * BT, s0 = blockIdx.y * BT, b = blockIdx.z;
    const int tid = threadIdx.x;
    const int lane = tid & 31, wave = tid >> 5;
    const int wm = wave >> 2, wn = wave & 3;       // 2 x 4 wave grid
    const int hl = lane >> 4, l15 = lane & 15;

    const f16* Ag = qnT + (size_t)q0 * CK;
    const f16* Bg = snT + ((size_t)b * HWP + s0) * CK;

    v8f acc[4][2] = {};   // [mi][ni]: wave covers 64(m) x 32(n)

    for (int k0 = 0; k0 < CK; k0 += BK) {
        __syncthreads();
        // Stage 128x64 f16 per operand: 16B vector load + 16B LDS store.
        #pragma unroll
        for (int v = 0; v < 4; ++v) {
            int idx = tid + v * 256;
            int row = idx >> 3, ko = (idx & 7) * 8;
            *(v8h*)&As[row * LDT + ko] = *(const v8h*)&Ag[(size_t)row * CK + k0 + ko];
            *(v8h*)&Bs[row * LDT + ko] = *(const v8h*)&Bg[(size_t)row * CK + k0 + ko];
        }
        if (k0 + BK < CK && tid < BT) {   // prefetch next chunk's rows
            __builtin_prefetch(&Ag[(size_t)tid * CK + k0 + BK], 0, 1);
            __builtin_prefetch(&Bg[(size_t)tid * CK + k0 + BK], 0, 1);
        }
        __syncthreads();

        #pragma unroll
        for (int kk = 0; kk < 2; ++kk) {
            v16h bf[2];
            #pragma unroll
            for (int ni = 0; ni < 2; ++ni)
                bf[ni] = load_b_frag(Bs + kk * 32, wn * 32 + ni * 16 + l15, hl);
            #pragma unroll
            for (int mi = 0; mi < 4; ++mi) {
                v16h a = load_a_frag(As + kk * 32, wm * 64 + mi * 16 + l15, hl);
                #pragma unroll
                for (int ni = 0; ni < 2; ++ni)
                    acc[mi][ni] = __builtin_amdgcn_wmma_f32_16x16x32_f16(
                        false, a, false, bf[ni],
                        (short)0, acc[mi][ni], false, false);
            }
        }
    }

    #pragma unroll
    for (int mi = 0; mi < 4; ++mi) {
        #pragma unroll
        for (int ni = 0; ni < 2; ++ni) {
            v8f res = acc[mi][ni] * 20.0f;             // TEMP
            int n = s0 + wn * 32 + ni * 16 + l15;
            if (n < HW) {
                #pragma unroll
                for (int r = 0; r < 8; ++r) {
                    int m = q0 + wm * 64 + mi * 16 + r + hl * 8;
                    if (m < HW) corr[((size_t)b * HW + m) * HW + n] = res[r];
                }
            }
        }
    }
}

// ---------------------------------------------------------------------------
// Row softmax; writes attn f16 into padded (b, HWP, HWP) layout:
// s in [HW,HWP) zero-filled (feeds AV K-sum), q rows >= HW left untouched.
__global__ __launch_bounds__(256)
void softmax_kernel(const float* __restrict__ corr, f16* __restrict__ attn) {
    __shared__ float red[256];
    const int row = blockIdx.x;            // b*HW + q
    const int b = row / HW, q = row - b * HW;
    const float* x = corr + (size_t)row * HW;
    f16* o = attn + ((size_t)b * HWP + q) * HWP;
    const int tid = threadIdx.x;

    float m = -1e30f;
    for (int i = tid; i < HW; i += 256) m = fmaxf(m, x[i]);
    red[tid] = m; __syncthreads();
    for (int s = 128; s > 0; s >>= 1) {
        if (tid < s) red[tid] = fmaxf(red[tid], red[tid + s]);
        __syncthreads();
    }
    m = red[0]; __syncthreads();

    float sum = 0.f;
    for (int i = tid; i < HW; i += 256) sum += __expf(x[i] - m);
    red[tid] = sum; __syncthreads();
    for (int s = 128; s > 0; s >>= 1) {
        if (tid < s) red[tid] += red[tid + s];
        __syncthreads();
    }
    float inv = 1.0f / red[0];

    for (int i = tid; i < HW; i += 256) o[i] = (f16)(__expf(x[i] - m) * inv);
    for (int i = HW + tid; i < HWP; i += 256) o[i] = (f16)0.f;  // zero s-pad
}

// ---------------------------------------------------------------------------
// out[c,q] = 0.5*f_q[c,q] + 0.25 * sum_{b,s} v[b,c,s]*attn[b,q,s]
// M=c (512), N=q, K=2*HWP (7424; shot-mean fused into K, padding zero).
__global__ __launch_bounds__(256)
void av_kernel(const f16* __restrict__ vbuf, const f16* __restrict__ attn,
               const float* __restrict__ f_q, float* __restrict__ out) {
    __shared__ f16 As[BT * LDT];   // [c_local][k]
    __shared__ f16 Bs[BT * LDT];   // [q_local][k]

    const int q0 = blockIdx.x * BT, c0 = blockIdx.y * BT;
    const int tid = threadIdx.x;
    const int lane = tid & 31, wave = tid >> 5;
    const int wm = wave >> 2, wn = wave & 3;
    const int hl = lane >> 4, l15 = lane & 15;

    v8f acc[4][2] = {};

    for (int k0 = 0; k0 < 2 * HWP; k0 += BK) {
        const int b = k0 >= HWP;            // HWP % BK == 0: never straddles
        const int sk = k0 - b * HWP;
        const f16* Ag = vbuf + ((size_t)b * 512 + c0) * HWP + sk;
        const f16* Bg = attn + ((size_t)b * HWP + q0) * HWP + sk;
        __syncthreads();
        #pragma unroll
        for (int v = 0; v < 4; ++v) {
            int idx = tid + v * 256;
            int row = idx >> 3, ko = (idx & 7) * 8;
            *(v8h*)&As[row * LDT + ko] = *(const v8h*)&Ag[(size_t)row * HWP + ko];
            *(v8h*)&Bs[row * LDT + ko] = *(const v8h*)&Bg[(size_t)row * HWP + ko];
        }
        __syncthreads();

        #pragma unroll
        for (int kk = 0; kk < 2; ++kk) {
            v16h bf[2];
            #pragma unroll
            for (int ni = 0; ni < 2; ++ni)
                bf[ni] = load_b_frag(Bs + kk * 32, wn * 32 + ni * 16 + l15, hl);
            #pragma unroll
            for (int mi = 0; mi < 4; ++mi) {
                v16h a = load_a_frag(As + kk * 32, wm * 64 + mi * 16 + l15, hl);
                #pragma unroll
                for (int ni = 0; ni < 2; ++ni)
                    acc[mi][ni] = __builtin_amdgcn_wmma_f32_16x16x32_f16(
                        false, a, false, bf[ni],
                        (short)0, acc[mi][ni], false, false);
            }
        }
    }

    #pragma unroll
    for (int mi = 0; mi < 4; ++mi) {
        #pragma unroll
        for (int ni = 0; ni < 2; ++ni) {
            int n = q0 + wn * 32 + ni * 16 + l15;
            if (n < HW) {
                #pragma unroll
                for (int r = 0; r < 8; ++r) {
                    int m = c0 + wm * 64 + mi * 16 + r + hl * 8;  // m < 512 always
                    out[(size_t)m * HW + n] =
                        0.5f * f_q[(size_t)m * HW + n] + 0.25f * acc[mi][ni][r];
                }
            }
        }
    }
}

// ---------------------------------------------------------------------------
extern "C" void kernel_launch(void* const* d_in, const int* in_sizes, int n_in,
                              void* d_out, int out_size, void* d_ws, size_t ws_size,
                              hipStream_t stream) {
    const float* fq3    = (const float*)d_in[0];  // (1,1024,60,60)
    const float* fq4    = (const float*)d_in[1];  // (1,2048,60,60)
    const float* fs3    = (const float*)d_in[2];  // (2,1024,60,60)
    const float* fs4    = (const float*)d_in[3];  // (2,2048,60,60)
    const float* f_q    = (const float*)d_in[4];  // (1,512,60,60)
    const float* f_s    = (const float*)d_in[5];  // (2,512,60,60)
    const float* w_chan = (const float*)d_in[6];  // (2,)
    float* out = (float*)d_out;

    char* ws = (char*)d_ws;
    size_t off = 0;
    auto take = [&](size_t bytes) {
        char* p = ws + off;
        off += (bytes + 255) & ~(size_t)255;
        return p;
    };
    f16*   qnT  = (f16*)  take((size_t)HWP * CK * sizeof(f16));       // [HWP][CK]
    f16*   snT  = (f16*)  take((size_t)2 * HWP * CK * sizeof(f16));   // [2][HWP][CK]
    f16*   vbuf = (f16*)  take((size_t)2 * 512 * HWP * sizeof(f16));
    f16*   attn = (f16*)  take((size_t)2 * HWP * HWP * sizeof(f16));
    float* corr = (float*)take((size_t)2 * HW * HW * sizeof(float));

    // 1) Normalize + transpose, concatenating pyramid levels along K.
    //    w_chan folded into the query operand.
    normalize_t_kernel<<<(HWP + 255) / 256, 256, 0, stream>>>(
        fq4, qnT, 2048, 0, 1, w_chan, 0);
    normalize_t_kernel<<<(HWP + 255) / 256, 256, 0, stream>>>(
        fq3, qnT, 1024, 2048, 1, w_chan, 1);
    normalize_t_kernel<<<(2 * HWP + 255) / 256, 256, 0, stream>>>(
        fs4, snT, 2048, 0, 2, w_chan, -1);
    normalize_t_kernel<<<(2 * HWP + 255) / 256, 256, 0, stream>>>(
        fs3, snT, 1024, 2048, 2, w_chan, -1);
    convert_pad_kernel<<<(2 * 512 * HWP + 255) / 256, 256, 0, stream>>>(
        f_s, vbuf, 2 * 512);

    // 2) Single fused correlation GEMM over K=3072 (WMMA).
    const int nt = HWP / BT;  // 29
    corr_kernel<<<dim3(nt, nt, 2), 256, 0, stream>>>(qnT, snT, corr);

    // 3) Row softmax -> f16 attention probabilities (zero-padded columns).
    softmax_kernel<<<2 * HW, 256, 0, stream>>>(corr, attn);

    // 4) attn @ V GEMM (WMMA), shot-mean + residual blend fused in epilogue.
    av_kernel<<<dim3(nt, 512 / BT), 256, 0, stream>>>(vbuf, attn, f_q, out);
}
